// HMCB_86706799771802
// MI455X (gfx1250) — matrix operand order repaired and checked by reading, not compiled, not run on hardware
//
#include <hip/hip_runtime.h>
#include <hip/hip_bf16.h>

// ---------------- problem constants (from reference) ----------------
#define L_TOT 4096   // H*W
#define HH    64
#define WW    64
#define DM    96     // D_MODEL
#define DI    192    // D_INNER
#define E2    384    // 2*D_INNER
#define NS    16     // D_STATE
#define DTR   6      // DT_RANK
#define KD    4      // directions
#define CPROJ 38     // DT_RANK + 2*D_STATE
#define CPAD  48     // padded to 3 WMMA m-tiles

typedef _Float16     v16h  __attribute__((ext_vector_type(16)));
typedef float        v8f   __attribute__((ext_vector_type(8)));
typedef unsigned int u32x4 __attribute__((ext_vector_type(4)));
typedef int          i32x8 __attribute__((ext_vector_type(8)));
typedef int          i32x4 __attribute__((ext_vector_type(4)));

// Direction index map: xs[k][d][l] == xc[d][dirmap(k,l)]
__device__ __forceinline__ int dirmap(int k, int l) {
    int j = (k >= 2) ? (L_TOT - 1 - l) : l;      // reversed directions
    if (k & 1) j = (j & 63) * 64 + (j >> 6);     // wh (transposed) directions
    return j;
}

// ---------------------------------------------------------------
// Kernel 1: in_proj  xz[e=384][l] = sum_c in_proj_w[e][c] * x[l][c]
// WMMA: A = W (384x96), B = x^T (96 x 4096). 24 x 256 tiles, K=3x32.
// ---------------------------------------------------------------
__global__ __launch_bounds__(256)
void k_inproj(const float* __restrict__ x, const float* __restrict__ w,
              float* __restrict__ xz) {
    int wave = (blockIdx.x * blockDim.x + threadIdx.x) >> 5;
    int lane = threadIdx.x & 31;
    const int NT = L_TOT / 16;                 // 256
    int mt = wave / NT, nt = wave % NT;
    if (mt >= E2 / 16) return;                 // 24 m-tiles
    int m0 = mt * 16, n0 = nt * 16;
    int m = lane & 15, khalf = (lane >> 4) << 3;
    v8f c = {};
    for (int kt = 0; kt < 3; ++kt) {
        int kb = kt * 32;
        v16h a, b;
#pragma unroll
        for (int i = 0; i < 16; ++i) {
            int kk = kb + khalf + ((i >= 8) ? 16 : 0) + (i & 7);
            a[i] = (_Float16)w[(m0 + m) * DM + kk];
        }
        int nc = n0 + (lane & 15);             // B: lane indexes N
#pragma unroll
        for (int i = 0; i < 16; ++i)
            b[i] = (_Float16)x[nc * DM + kb + khalf + ((i >= 8) ? 16 : 0) + (i & 7)];
        c = __builtin_amdgcn_wmma_f32_16x16x32_f16(false, a, false, b,
                                                   (short)0, c, false, false);
    }
    int n = lane & 15, mb = (lane >> 4) << 3;
#pragma unroll
    for (int v = 0; v < 8; ++v)
        xz[(m0 + mb + v) * L_TOT + (n0 + n)] = c[v];
}

// ---------------------------------------------------------------
// Kernel 2: depthwise 3x3 conv + bias + SiLU on first 192 rows of xz
// ---------------------------------------------------------------
__global__ __launch_bounds__(256)
void k_conv(const float* __restrict__ xz, const float* __restrict__ cw,
            const float* __restrict__ cb, float* __restrict__ xc) {
    int idx = blockIdx.x * blockDim.x + threadIdx.x;
    if (idx >= DI * L_TOT) return;
    int d = idx / L_TOT, l = idx - d * L_TOT;
    int y = l >> 6, xq = l & 63;
    float acc = cb[d];
#pragma unroll
    for (int dy = -1; dy <= 1; ++dy)
#pragma unroll
        for (int dx = -1; dx <= 1; ++dx) {
            int yy = y + dy, xx = xq + dx;
            if (yy >= 0 && yy < HH && xx >= 0 && xx < WW)
                acc += cw[d * 9 + (dy + 1) * 3 + (dx + 1)] *
                       xz[d * L_TOT + yy * WW + xx];
        }
    xc[idx] = acc / (1.f + __expf(-acc));      // SiLU
}

// ---------------------------------------------------------------
// Kernel 3: x_dbl[k][c][l] = sum_d x_proj_w[k][c][d] * xs[k][d][l]
// M padded 38->48 (3 tiles), K = 192 = 6x32, per direction k.
// ---------------------------------------------------------------
__global__ __launch_bounds__(256)
void k_xdbl(const float* __restrict__ xc, const float* __restrict__ xpw,
            float* __restrict__ xdbl) {
    int wave = (blockIdx.x * blockDim.x + threadIdx.x) >> 5;
    int lane = threadIdx.x & 31;
    const int NT = L_TOT / 16;
    const int PER_K = (CPAD / 16) * NT;        // 3*256
    int k = wave / PER_K;
    if (k >= KD) return;
    int rem = wave - k * PER_K;
    int mt = rem / NT, nt = rem - mt * NT;
    int m0 = mt * 16, n0 = nt * 16;
    int m = lane & 15, khalf = (lane >> 4) << 3;
    v8f c = {};
    for (int kt = 0; kt < 6; ++kt) {
        int kb = kt * 32;
        v16h a, b;
        int row = m0 + m;
#pragma unroll
        for (int i = 0; i < 16; ++i) {
            int kk = kb + khalf + ((i >= 8) ? 16 : 0) + (i & 7);
            a[i] = (row < CPROJ) ? (_Float16)xpw[(k * CPROJ + row) * DI + kk]
                                 : (_Float16)0.f;
        }
        int nc = n0 + (lane & 15);
#pragma unroll
        for (int i = 0; i < 16; ++i)
            b[i] = (_Float16)xc[(kb + khalf + ((i >= 8) ? 16 : 0) + (i & 7)) * L_TOT + dirmap(k, nc)];
        c = __builtin_amdgcn_wmma_f32_16x16x32_f16(false, a, false, b,
                                                   (short)0, c, false, false);
    }
    int n = lane & 15, mb = (lane >> 4) << 3;
#pragma unroll
    for (int v = 0; v < 8; ++v)
        xdbl[(k * CPAD + m0 + mb + v) * L_TOT + (n0 + n)] = c[v];
}

// ---------------------------------------------------------------
// Kernel 4: dt projection  dtp[k][d][l] = sum_r dt_w[k][d][r]*dts[k][r][l]
// K = 6 padded to one 32-step WMMA.
// ---------------------------------------------------------------
__global__ __launch_bounds__(256)
void k_dtproj(const float* __restrict__ xdbl, const float* __restrict__ dtw,
              float* __restrict__ dtp) {
    int wave = (blockIdx.x * blockDim.x + threadIdx.x) >> 5;
    int lane = threadIdx.x & 31;
    const int NT = L_TOT / 16;
    const int PER_K = (DI / 16) * NT;          // 12*256
    int k = wave / PER_K;
    if (k >= KD) return;
    int rem = wave - k * PER_K;
    int mt = rem / NT, nt = rem - mt * NT;
    int m0 = mt * 16, n0 = nt * 16;
    int m = lane & 15, khalf = (lane >> 4) << 3;
    v16h a, b;
#pragma unroll
    for (int i = 0; i < 16; ++i) {
        int kk = khalf + ((i >= 8) ? 16 : 0) + (i & 7);
        a[i] = (kk < DTR) ? (_Float16)dtw[(k * DI + m0 + m) * DTR + kk]
                          : (_Float16)0.f;
    }
    int nc = n0 + (lane & 15);
#pragma unroll
    for (int i = 0; i < 16; ++i) {
        int kr = khalf + ((i >= 8) ? 16 : 0) + (i & 7);
        b[i] = (kr < DTR) ? (_Float16)xdbl[(k * CPAD + kr) * L_TOT + nc]
                          : (_Float16)0.f;
    }
    v8f c = {};
    c = __builtin_amdgcn_wmma_f32_16x16x32_f16(false, a, false, b,
                                               (short)0, c, false, false);
    int n = lane & 15, mb = (lane >> 4) << 3;
#pragma unroll
    for (int v = 0; v < 8; ++v)
        dtp[(k * DI + m0 + mb + v) * L_TOT + (n0 + n)] = c[v];
}

// ---------------------------------------------------------------
// Kernel 5: selective scan. One wave per (k, 2 channels):
// lane = d_local*16 + n ; each lane carries one h state in a register.
// State reduction across 16 lanes via shfl_xor (ds-permute path).
// ---------------------------------------------------------------
__global__ __launch_bounds__(128)
void k_scan(const float* __restrict__ xc, const float* __restrict__ xdbl,
            const float* __restrict__ dtp, const float* __restrict__ dtb,
            const float* __restrict__ alogs, const float* __restrict__ dsv,
            float* __restrict__ yk) {
    int wave = (blockIdx.x * blockDim.x + threadIdx.x) >> 5;
    int lane = threadIdx.x & 31;
    int k = wave / (DI / 2);
    if (k >= KD) return;
    int dpair = wave - k * (DI / 2);
    int dl = lane >> 4, n = lane & 15;
    int d = dpair * 2 + dl;

    float Aval = -__expf(alogs[(k * DI + d) * NS + n]);
    float dtbv = dtb[k * DI + d];
    float Dsv  = dsv[k * DI + d];

    const float* dtrow = dtp  + (k * DI + d) * L_TOT;
    const float* Brow  = xdbl + (k * CPAD + DTR + n) * L_TOT;
    const float* Crow  = xdbl + (k * CPAD + DTR + NS + n) * L_TOT;
    const float* xrow  = xc   + d * L_TOT;
    float*       yrow  = yk   + (k * DI + d) * L_TOT;

    float h = 0.f;
    for (int l = 0; l < L_TOT; ++l) {
        if ((l & 31) == 0) {                    // stream-ahead prefetch
            __builtin_prefetch(dtrow + l + 128, 0, 1);
            __builtin_prefetch(Brow  + l + 128, 0, 1);
            __builtin_prefetch(Crow  + l + 128, 0, 1);
        }
        int j = dirmap(k, l);
        float dtv = dtrow[l] + dtbv;
        float delta = (dtv > 20.f) ? dtv : __logf(1.f + __expf(dtv));
        float xv = xrow[j];
        float dA = __expf(delta * Aval);
        h = dA * h + delta * Brow[l] * xv;
        float contrib = h * Crow[l];
        contrib += __shfl_xor(contrib, 1, 32);
        contrib += __shfl_xor(contrib, 2, 32);
        contrib += __shfl_xor(contrib, 4, 32);
        contrib += __shfl_xor(contrib, 8, 32);
        if (n == 0) yrow[l] = contrib + Dsv * xv;
    }
}

// ---------------------------------------------------------------
// Kernel 6: combine 4 directions + LayerNorm over d + SiLU(z) gate
// One block per spatial position l (192 threads = 6 waves).
// ---------------------------------------------------------------
__global__ __launch_bounds__(192)
void k_combine(const float* __restrict__ yk, const float* __restrict__ xz,
               const float* __restrict__ lng, const float* __restrict__ lnb,
               float* __restrict__ yg) {
    int l = blockIdx.x;
    int d = threadIdx.x;
    int tl = (l & 63) * 64 + (l >> 6);         // hw<->wh transpose of l

    float v = yk[(0 * DI + d) * L_TOT + l]
            + yk[(2 * DI + d) * L_TOT + (L_TOT - 1 - l)]
            + yk[(1 * DI + d) * L_TOT + tl]
            + yk[(3 * DI + d) * L_TOT + (L_TOT - 1 - tl)];

    __shared__ float red[24];
    float s = v, s2 = v * v;
#pragma unroll
    for (int o = 16; o > 0; o >>= 1) {
        s  += __shfl_down(s,  o, 32);
        s2 += __shfl_down(s2, o, 32);
    }
    int wid = threadIdx.x >> 5;
    if ((threadIdx.x & 31) == 0) { red[wid] = s; red[8 + wid] = s2; }
    __syncthreads();
    if (threadIdx.x == 0) {
        float ts = 0.f, ts2 = 0.f;
#pragma unroll
        for (int i = 0; i < 6; ++i) { ts += red[i]; ts2 += red[8 + i]; }
        red[16] = ts / (float)DI;
        red[17] = ts2 / (float)DI;
    }
    __syncthreads();
    float mu = red[16];
    float var = red[17] - mu * mu;
    float nv = (v - mu) * rsqrtf(var + 1e-5f) * lng[d] + lnb[d];
    float zv = xz[(DI + d) * L_TOT + l];       // z half of in_proj
    nv *= zv / (1.f + __expf(-zv));            // * silu(z)
    yg[d * L_TOT + l] = nv;
}

// ---------------------------------------------------------------
// Kernel 7: out_proj  out[l][c] = sum_d out_proj_w[c][d] * yg[d][l]
// B tiles (32 K-rows x 16 l-cols, f32) staged in LDS via the Tensor
// Data Mover (TENSOR_LOAD_TO_LDS + s_wait_tensorcnt), then WMMA.
// M = 96 (6 tiles), K = 192 = 6x32, output stored [L, 96].
// ---------------------------------------------------------------
__global__ __launch_bounds__(256)
void k_outproj(const float* __restrict__ yg, const float* __restrict__ ow,
               float* __restrict__ out) {
    __shared__ float btile[8][32 * 16];        // 2KB per wave, 16KB per block
    int wave_l = threadIdx.x >> 5;
    int wave = (blockIdx.x * blockDim.x + threadIdx.x) >> 5;
    int lane = threadIdx.x & 31;
    const int NT = L_TOT / 16;
    int mt = wave / NT;                        // grid sized so mt < 6 always
    int nt = wave - mt * NT;
    int m0 = mt * 16, n0 = nt * 16;
    int m = lane & 15, khalf = (lane >> 4) << 3;

    // LDS byte offset of this wave's staging slot (flat addr low 32 bits)
    unsigned int lds_base =
        (unsigned int)(unsigned long long)&btile[wave_l][0];

    v8f c = {};
    for (int kt = 0; kt < 6; ++kt) {
        int kb = kt * 32;
        // ---- TDM descriptor: 2D tensor yg[192][4096] f32,
        //      tile = 32 rows x 16 cols starting at (kb, n0) ----
        unsigned long long gaddr =
            (unsigned long long)(const void*)(yg + (size_t)kb * L_TOT + n0);
        u32x4 g0 = {};
        g0[0] = 1u;                                        // count=1 (valid)
        g0[1] = lds_base;                                  // lds_addr
        g0[2] = (unsigned int)gaddr;                       // global_addr[31:0]
        g0[3] = (unsigned int)(gaddr >> 32) | (2u << 30);  // addr[56:32]|type=2
        i32x8 g1 = {};
        g1[0] = (2 << 16);                  // data_size=2 (4B), wg_mask=0
        g1[1] = (int)(L_TOT << 16);         // tensor_dim0[15:0]=4096
        g1[2] = (DI << 16);                 // dim0 hi=0 | tensor_dim1[15:0]=192
        g1[3] = (16 << 16);                 // dim1 hi=0 | tile_dim0=16
        g1[4] = 32;                         // tile_dim1=32, tile_dim2=0
        g1[5] = (int)L_TOT;                 // tensor_dim0_stride[31:0]=4096
        g1[6] = 0;                          // stride hi | dim1_stride lo
        g1[7] = 0;
        i32x4 z4 = {};
        i32x8 z8 = {};
        __builtin_amdgcn_tensor_load_to_lds(g0, g1, z4, z4, z8, 0);
        __builtin_amdgcn_s_wait_tensorcnt(0);

        v16h a, b;
#pragma unroll
        for (int i = 0; i < 16; ++i) {
            int kk = kb + khalf + ((i >= 8) ? 16 : 0) + (i & 7);
            a[i] = (_Float16)ow[(m0 + m) * DI + kk];
        }
#pragma unroll
        for (int i = 0; i < 16; ++i)        // B: lane%16 = N col, i = K row
            b[i] = (_Float16)btile[wave_l][(khalf + ((i >= 8) ? 16 : 0) + (i & 7)) * 16 + (lane & 15)];
        c = __builtin_amdgcn_wmma_f32_16x16x32_f16(false, a, false, b,
                                                   (short)0, c, false, false);
    }
    int n = lane & 15, mb = (lane >> 4) << 3;
#pragma unroll
    for (int v = 0; v < 8; ++v)
        out[(n0 + n) * DM + (m0 + mb + v)] = c[v];   // [l][c] layout
}

// ---------------------------------------------------------------
extern "C" void kernel_launch(void* const* d_in, const int* in_sizes, int n_in,
                              void* d_out, int out_size, void* d_ws, size_t ws_size,
                              hipStream_t stream) {
    (void)in_sizes; (void)n_in; (void)out_size; (void)ws_size;
    const float* x     = (const float*)d_in[0];   // [4096,96]
    const float* ipw   = (const float*)d_in[1];   // [384,96]
    const float* cw    = (const float*)d_in[2];   // [192,1,3,3]
    const float* cb    = (const float*)d_in[3];   // [192]
    const float* xpw   = (const float*)d_in[4];   // [4,38,192]
    const float* dtw   = (const float*)d_in[5];   // [4,192,6]
    const float* dtb   = (const float*)d_in[6];   // [4,192]
    const float* alogs = (const float*)d_in[7];   // [768,16]
    const float* dsv   = (const float*)d_in[8];   // [768]
    const float* lng   = (const float*)d_in[9];   // [192]
    const float* lnb   = (const float*)d_in[10];  // [192]
    const float* opw   = (const float*)d_in[11];  // [96,192]
    float* out = (float*)d_out;

    char* ws = (char*)d_ws;
    float* xz   = (float*)(ws);                                  // [384][4096]
    float* xc   = (float*)(ws + (size_t)E2 * L_TOT * 4);         // [192][4096]
    float* xdbl = (float*)((char*)xc   + (size_t)DI * L_TOT * 4);        // [4][48][4096]
    float* dtp  = (float*)((char*)xdbl + (size_t)KD * CPAD * L_TOT * 4); // [4][192][4096]
    float* yk   = (float*)((char*)dtp  + (size_t)KD * DI * L_TOT * 4);   // [4][192][4096]
    float* yg   = (float*)((char*)yk   + (size_t)KD * DI * L_TOT * 4);   // [192][4096]

    // 1) in_proj: 24*256 wave-tiles -> 768 blocks of 256
    k_inproj<<<768, 256, 0, stream>>>(x, ipw, xz);
    // 2) depthwise conv + SiLU
    k_conv<<<(DI * L_TOT + 255) / 256, 256, 0, stream>>>(xz, cw, cb, xc);
    // 3) x_dbl projection: 4*3*256 wave-tiles -> 384 blocks
    k_xdbl<<<384, 256, 0, stream>>>(xc, xpw, xdbl);
    // 4) dt projection: 4*12*256 wave-tiles -> 1536 blocks
    k_dtproj<<<1536, 256, 0, stream>>>(xdbl, dtw, dtp);
    // 5) selective scan: 384 waves -> 96 blocks of 128
    k_scan<<<96, 128, 0, stream>>>(xc, xdbl, dtp, dtb, alogs, dsv, yk);
    // 6) combine + LN + gate: one block per l
    k_combine<<<L_TOT, DI, 0, stream>>>(yk, xz, lng, lnb, yg);
    // 7) out_proj (TDM-staged B tiles): 6*256 wave-tiles -> 192 blocks
    k_outproj<<<192, 256, 0, stream>>>(yg, opw, out);
}
